// Schnax_6262062317919
// MI455X (gfx1250) — matrix operand order repaired and checked by reading, not compiled, not run on hardware
//
#include <hip/hip_runtime.h>
#include <hip/hip_bf16.h>

// ---- types for WMMA fragments (CDNA5 wave32) ----
typedef __attribute__((ext_vector_type(16))) _Float16 v16h;
typedef __attribute__((ext_vector_type(8)))  _Float16 v8h;
typedef __attribute__((ext_vector_type(8)))  float    v8f;

#define NG        25      // gaussians
#define NF        128     // filters / atom basis
#define KNB       48      // neighbors per atom
#define RC        5.0f    // cutoff
#define PADR      136     // padded row length (halves) for LDS staging, mult of 8

// shifted softplus: log(1 + e^x) - log(2)
__device__ __forceinline__ float ssp(float x) {
    return fmaxf(x, 0.0f) + log1pf(__expf(-fabsf(x))) - 0.69314718056f;
}

// -------------------------------------------------------------------------
// Pack a [Krows, 128] f32 weight matrix into f16 WMMA B-fragments.
// Fragment (kt, nt) holds B[kt*32 .. kt*32+31][nt*16 .. nt*16+15].
// v_wmma_f32_16x16x32_f16 B layout: lane l holds column n = l%16 and 16
// halves; half i is K = (i/8)*16 + (l/16)*8 + (i%8) within the tile.
// Packed so each lane loads its 16 halves from [frag*512 + lane*16].
// -------------------------------------------------------------------------
__global__ void pack_b_frags(const float* __restrict__ src,
                             _Float16* __restrict__ dst,
                             int Krows, int Ktiles) {
    int tid = blockIdx.x * 256 + threadIdx.x;
    int total = Ktiles * 8 * 512;
    if (tid >= total) return;
    int i    = tid & 15;
    int lane = (tid >> 4) & 31;
    int nt   = (tid >> 9) & 7;
    int kt   = tid >> 12;
    int k = kt * 32 + (i >> 3) * 16 + (lane >> 4) * 8 + (i & 7);
    int n = nt * 16 + (lane & 15);
    float v = (k < Krows) ? src[k * NF + n] : 0.0f;
    dst[tid] = (_Float16)v;
}

// Build an A-fragment (16x32 f16) from a row-major f16 LDS buffer [16][PADR].
// A layout: lane l holds M = l%16; half i: K = (i/8)*16 + (l/16)*8 + (i%8)
// -> two contiguous 8-half (16B) LDS reads per kt.
__device__ __forceinline__ v16h load_a_frag(const _Float16* h, int mrow, int hi, int kt) {
    v8h lo = *(const v8h*)(h + mrow * PADR + kt * 32 + hi * 8);
    v8h hg = *(const v8h*)(h + mrow * PADR + kt * 32 + 16 + hi * 8);
    v16h a;
#pragma unroll
    for (int j = 0; j < 8; ++j) { a[j] = lo[j]; a[8 + j] = hg[j]; }
    return a;
}

// Cooperative gaussian A-fragment image: 512 halves, 2 exps per thread.
__device__ __forceinline__ void build_gfrag(_Float16* __restrict__ gbuf,
                                            const float* __restrict__ sdr,
                                            int k, int t, float width, float coeff) {
#pragma unroll
    for (int q = 0; q < 2; ++q) {
        int e  = t + q * 256;
        int ln = e >> 4, ii = e & 15;
        int M  = ln & 15, h2 = ln >> 4;
        int j  = (ii >> 3) * 16 + h2 * 8 + (ii & 7);   // gaussian idx 0..31
        float d  = sdr[M * KNB + k];
        float dd = d - (float)j * width;
        gbuf[e] = (j < NG) ? (_Float16)__expf(coeff * dd * dd)
                           : (_Float16)0.0f;
    }
}

// -------------------------------------------------------------------------
// Kernel A: y = embed[Z] @ in2f_w   (f32 out, WMMA f16 inputs / f32 accum)
// One WG = 16 atoms, 8 waves; wave w owns output columns [w*16, w*16+16).
// -------------------------------------------------------------------------
__global__ __launch_bounds__(256) void project_kernel(
    const float* __restrict__ embed, const int* __restrict__ Z,
    const _Float16* __restrict__ pin2f, float* __restrict__ y, int N)
{
    __shared__ _Float16 xr[16 * PADR];
    int t = threadIdx.x;
    int atom0 = blockIdx.x * 16;
    bool full = (atom0 + 16 <= N);

    // stage 16 embedding rows (f32 -> f16) into LDS, vectorized
    int m  = t >> 4;           // 0..15
    int c0 = (t & 15) * 8;     // 8 cols per thread
    int am = atom0 + m; if (am >= N) am = N - 1;
    int z = Z[am];
    const float4* er4 = (const float4*)(embed + (size_t)z * NF + c0);
    float4 e0 = er4[0], e1 = er4[1];
    v8h xh;
    xh[0] = (_Float16)e0.x; xh[1] = (_Float16)e0.y;
    xh[2] = (_Float16)e0.z; xh[3] = (_Float16)e0.w;
    xh[4] = (_Float16)e1.x; xh[5] = (_Float16)e1.y;
    xh[6] = (_Float16)e1.z; xh[7] = (_Float16)e1.w;
    *(v8h*)(xr + m * PADR + c0) = xh;
    __syncthreads();

    int wave = t >> 5, lane = t & 31;
    int mrow = lane & 15, hi = lane >> 4;
    v8f c = {};
#pragma unroll
    for (int kt = 0; kt < 4; ++kt) {
        v16h a = load_a_frag(xr, mrow, hi, kt);
        v16h b = *(const v16h*)(pin2f + (kt * 8 + wave) * 512 + lane * 16);
        c = __builtin_amdgcn_wmma_f32_16x16x32_f16(false, a, false, b,
                                                   (short)0, c, false, false);
    }
    int col = wave * 16 + mrow;
    float* yp = y + (size_t)(atom0 + 8 * hi) * NF + col;
    if (full) {
#pragma unroll
        for (int r = 0; r < 8; ++r) yp[(size_t)r * NF] = c[r];
    } else {
#pragma unroll
        for (int r = 0; r < 8; ++r)
            if (atom0 + 8 * hi + r < N) yp[(size_t)r * NF] = c[r];
    }
}

// -------------------------------------------------------------------------
// Kernel B: fused filter network + cfconv aggregation + output network.
// One WG = 16 atoms.  Software-pipelined hot loop, ONE barrier / neighbor:
//   iter k: GEMM1 reads gfrag[k%2] -> ssp -> h1[k%2]; build gfrag[(k+1)%2];
//           barrier; y-gather loads issued; GEMM2 reads h1[k%2]; accumulate.
// All cross-wave-redundant scalar work (exp/cos/masks/indices) is staged or
// computed cooperatively; f1/f2 B-fragments stay in registers for all 48
// iterations.  Epilogue: 4x WMMA (o1) -> ssp -> 4x WMMA (o2) -> residual.
// -------------------------------------------------------------------------
__global__ __launch_bounds__(256) void schnet_main_kernel(
    const float* __restrict__ dR, const int* __restrict__ Z,
    const int* __restrict__ nbr, const float* __restrict__ embed,
    const _Float16* __restrict__ pf1, const _Float16* __restrict__ pf2,
    const _Float16* __restrict__ po1, const _Float16* __restrict__ po2,
    const float* __restrict__ f1_b, const float* __restrict__ f2_b,
    const float* __restrict__ o1_b, const float* __restrict__ o2_b,
    const float* __restrict__ y, float* __restrict__ out, int N)
{
    __shared__ _Float16 h1buf[2][16 * PADR];  // double-buffered activations
    __shared__ _Float16 gbuf[2][512];         // double-buffered gaussian image
    __shared__ float    sdr[16 * KNB];        // dR rows for this tile
    __shared__ float    sfac[KNB * 16];       // cut * mask, [k][m]
    __shared__ int      sidx[KNB * 16];       // safe gather index, [k][m]
    __shared__ int      szz[16];              // Z for this tile

    const float width = RC / (float)(NG - 1);
    const float coeff = -0.5f / (width * width);

    int t = threadIdx.x;
    int atom0 = blockIdx.x * 16;
    bool full = (atom0 + 16 <= N);
    int wave = t >> 5, lane = t & 31;
    int mrow = lane & 15, hi = lane >> 4;
    int col = wave * 16 + mrow;

    // ---- prologue: stage per-tile data (768 = 3*256 pair slots) ----
#pragma unroll
    for (int q = 0; q < 3; ++q) {
        int p = t + q * 256;
        int m = p / KNB, k = p % KNB;
        int a2 = atom0 + m; if (a2 >= N) a2 = N - 1;
        float d = dR[(size_t)a2 * KNB + k];      // coalesced row reads
        int   id = nbr[(size_t)a2 * KNB + k];
        float cut = (d < RC) ? 0.5f * (__cosf(d * (3.14159265358979f / RC)) + 1.0f)
                             : 0.0f;
        sdr[m * KNB + k]  = d;
        sfac[k * 16 + m]  = (id < N) ? cut : 0.0f;
        sidx[k * 16 + m]  = (id < N) ? id : 0;
    }
    if (t < 16) {
        int a2 = atom0 + t; if (a2 >= N) a2 = N - 1;
        szz[t] = Z[a2];
    }

    float b1  = f1_b[col];
    float b2  = f2_b[col];
    float bo1 = o1_b[col];
    float bo2 = o2_b[col];

    // loop-invariant B fragments in registers
    v16h B1 = *(const v16h*)(pf1 + wave * 512 + lane * 16);
    v16h B2[4];
#pragma unroll
    for (int kt = 0; kt < 4; ++kt)
        B2[kt] = *(const v16h*)(pf2 + (kt * 8 + wave) * 512 + lane * 16);

    __syncthreads();   // prologue staging visible (sdr needed below)

    build_gfrag(gbuf[0], sdr, 0, t, width, coeff);   // prime the pipeline
    __syncthreads();

    v8f agg = {};

    for (int k = 0; k < KNB; ++k) {
        const _Float16* gb = gbuf[k & 1];
        _Float16*       hb = h1buf[k & 1];

        // ---- GEMM1: h = g @ f1_w + f1_b ----
        v16h ga = *(const v16h*)(gb + lane * 16);    // broadcast LDS read
        v8f hc;
#pragma unroll
        for (int r = 0; r < 8; ++r) hc[r] = b1;
        hc = __builtin_amdgcn_wmma_f32_16x16x32_f16(false, ga, false, B1,
                                                    (short)0, hc, false, false);
        // ssp + stage h1 (f16, row-major) into LDS
#pragma unroll
        for (int r = 0; r < 8; ++r)
            hb[(r + 8 * hi) * PADR + col] = (_Float16)ssp(hc[r]);

        // build next iteration's gaussian image into the other buffer
        if (k + 1 < KNB)
            build_gfrag(gbuf[(k + 1) & 1], sdr, k + 1, t, width, coeff);

        __syncthreads();   // h1(k) + gfrag(k+1) produced; prior readers done

        // ---- y-gather loads first: overlap L2 latency with WMMA chain ----
        const float* fk = sfac + k * 16;
        const int*   ik = sidx + k * 16;
        float yv[8];
#pragma unroll
        for (int r = 0; r < 8; ++r)
            yv[r] = y[(size_t)ik[r + 8 * hi] * NF + col];

        // ---- GEMM2: W = h1 @ f2_w + f2_b ----
        v8f wc;
#pragma unroll
        for (int r = 0; r < 8; ++r) wc[r] = b2;
#pragma unroll
        for (int kt = 0; kt < 4; ++kt) {
            v16h a = load_a_frag(hb, mrow, hi, kt);
            wc = __builtin_amdgcn_wmma_f32_16x16x32_f16(false, a, false, B2[kt],
                                                        (short)0, wc, false, false);
        }
        // ---- fused cutoff*mask scaling + aggregation ----
#pragma unroll
        for (int r = 0; r < 8; ++r)
            agg[r] += wc[r] * fk[r + 8 * hi] * yv[r];
    }

    // ---- output network: v = ssp(agg@o1 + b) @ o2 + b ; out = x + v ----
    __syncthreads();
#pragma unroll
    for (int r = 0; r < 8; ++r)
        h1buf[0][(r + 8 * hi) * PADR + col] = (_Float16)agg[r];
    __syncthreads();

    v8f c3;
#pragma unroll
    for (int r = 0; r < 8; ++r) c3[r] = bo1;
#pragma unroll
    for (int kt = 0; kt < 4; ++kt) {
        v16h a = load_a_frag(h1buf[0], mrow, hi, kt);
        v16h b = *(const v16h*)(po1 + (kt * 8 + wave) * 512 + lane * 16);
        c3 = __builtin_amdgcn_wmma_f32_16x16x32_f16(false, a, false, b,
                                                    (short)0, c3, false, false);
    }
    // write ssp(h2) into the OTHER buffer: no barrier needed before the write
#pragma unroll
    for (int r = 0; r < 8; ++r)
        h1buf[1][(r + 8 * hi) * PADR + col] = (_Float16)ssp(c3[r]);
    __syncthreads();

    v8f c4;
#pragma unroll
    for (int r = 0; r < 8; ++r) c4[r] = bo2;
#pragma unroll
    for (int kt = 0; kt < 4; ++kt) {
        v16h a = load_a_frag(h1buf[1], mrow, hi, kt);
        v16h b = *(const v16h*)(po2 + (kt * 8 + wave) * 512 + lane * 16);
        c4 = __builtin_amdgcn_wmma_f32_16x16x32_f16(false, a, false, b,
                                                    (short)0, c4, false, false);
    }

    float* op = out + (size_t)(atom0 + 8 * hi) * NF + col;
    if (full) {
#pragma unroll
        for (int r = 0; r < 8; ++r) {
            int z = szz[r + 8 * hi];
            op[(size_t)r * NF] = c4[r] + embed[(size_t)z * NF + col];
        }
    } else {
#pragma unroll
        for (int r = 0; r < 8; ++r) {
            if (atom0 + 8 * hi + r < N) {
                int z = szz[r + 8 * hi];
                op[(size_t)r * NF] = c4[r] + embed[(size_t)z * NF + col];
            }
        }
    }
}

// -------------------------------------------------------------------------
extern "C" void kernel_launch(void* const* d_in, const int* in_sizes, int n_in,
                              void* d_out, int out_size, void* d_ws, size_t ws_size,
                              hipStream_t stream) {
    const float* dR     = (const float*)d_in[0];
    const int*   Z      = (const int*)  d_in[1];
    const int*   nbr    = (const int*)  d_in[2];
    const float* embed  = (const float*)d_in[3];
    const float* in2f_w = (const float*)d_in[4];
    const float* f1_w   = (const float*)d_in[5];
    const float* f1_b   = (const float*)d_in[6];
    const float* f2_w   = (const float*)d_in[7];
    const float* f2_b   = (const float*)d_in[8];
    const float* o1_w   = (const float*)d_in[9];
    const float* o1_b   = (const float*)d_in[10];
    const float* o2_w   = (const float*)d_in[11];
    const float* o2_b   = (const float*)d_in[12];
    float* out = (float*)d_out;

    int N = in_sizes[1];           // number of atoms (20000)

    // workspace layout: y [N,128] f32, then packed f16 weight fragments
    float* y = (float*)d_ws;
    char* base = (char*)d_ws + (size_t)N * NF * sizeof(float);
    _Float16* pf1   = (_Float16*)(base);                 //  8 KB (1 K-tile)
    _Float16* pf2   = (_Float16*)(base + 8192);          // 32 KB
    _Float16* pin2f = (_Float16*)(base + 8192 + 32768);  // 32 KB
    _Float16* po1   = (_Float16*)(base + 8192 + 2 * 32768);
    _Float16* po2   = (_Float16*)(base + 8192 + 3 * 32768);

    pack_b_frags<<<16, 256, 0, stream>>>(f1_w,   pf1,   NG,  1);
    pack_b_frags<<<64, 256, 0, stream>>>(f2_w,   pf2,   NF,  4);
    pack_b_frags<<<64, 256, 0, stream>>>(in2f_w, pin2f, NF,  4);
    pack_b_frags<<<64, 256, 0, stream>>>(o1_w,   po1,   NF,  4);
    pack_b_frags<<<64, 256, 0, stream>>>(o2_w,   po2,   NF,  4);

    int tiles = (N + 15) / 16;
    project_kernel<<<tiles, 256, 0, stream>>>(embed, Z, pin2f, y, N);
    schnet_main_kernel<<<tiles, 256, 0, stream>>>(
        dR, Z, nbr, embed, pf1, pf2, po1, po2,
        f1_b, f2_b, o1_b, o2_b, y, out, N);
}
